// AttentionBlock_12154757448145
// MI455X (gfx1250) — compile-verified, matrix operand order checked by reading, not run on hardware
//
#include <hip/hip_runtime.h>

typedef __attribute__((ext_vector_type(16))) __bf16 v16bf;
typedef __attribute__((ext_vector_type(8)))  __bf16 v8bf;
typedef __attribute__((ext_vector_type(8)))  float  v8f;
typedef __attribute__((ext_vector_type(4)))  unsigned int u32x4;
typedef __attribute__((ext_vector_type(8)))  int i32x8;
typedef __attribute__((ext_vector_type(4)))  int i32x4;

#define B_  8
#define C_  512
#define S_  1024
#define NH  8
#define HD  64

#if __has_builtin(__builtin_amdgcn_tensor_load_to_lds) && \
    __has_builtin(__builtin_amdgcn_s_wait_tensorcnt)
#define USE_TDM 1
#endif

// ---- fragment helpers -------------------------------------------------------
// Row-major 16-bit M x K tile -> WMMA A/B fragment for lane L:
//   row/col = L%16, khalf = L/16
//   VGPR0..3 hold K = khalf*8..+7 ; VGPR4..7 hold K = 16+khalf*8..+7
// i.e. two 16-byte loads from (base + (L%16)*ld + khalf*8) and +16 elements.
__device__ __forceinline__ v16bf load_bf16_frag(const __bf16* p) {
    v8bf lo = *(const v8bf*)p;
    v8bf hi = *(const v8bf*)(p + 16);
    v16bf r;
#pragma unroll
    for (int i = 0; i < 8; ++i) { r[i] = lo[i]; r[i + 8] = hi[i]; }
    return r;
}

__device__ __forceinline__ v8f wmma_bf16(v16bf a, v16bf b, v8f c) {
    return __builtin_amdgcn_wmma_f32_16x16x32_bf16(
        false, a, false, b, (short)0, c, false, false);
}

#ifdef USE_TDM
__device__ __forceinline__ unsigned lds_off(const void* p) {
    return (unsigned)(unsigned long long)
        (__attribute__((address_space(3))) const void*)p;
}

// 2D TDM tile load: width elems (2B each) x rows, row stride in elems,
// LDS padding (pad_interval/pad_amount codes per D# group1).
__device__ __forceinline__ void tdm_load_2d(unsigned lds_addr, const void* gp,
                                            unsigned width, unsigned rows,
                                            unsigned stride,
                                            unsigned pad_int_code,
                                            unsigned pad_amt_code) {
    unsigned long long ga = (unsigned long long)gp;
    u32x4 g0;
    g0[0] = 1u;                                   // count=1, user descriptor
    g0[1] = lds_addr;                             // lds_addr (bytes)
    g0[2] = (unsigned)ga;                         // global_addr[31:0]
    g0[3] = ((unsigned)(ga >> 32) & 0x01FFFFFFu)  // global_addr[56:32]
            | (2u << 30);                         // type = 2 ("image")
    i32x8 g1;
    g1[0] = (int)((1u << 16)                      // data_size = 2 bytes
                  | (1u << 20)                    // pad_enable
                  | (pad_int_code << 22)
                  | (pad_amt_code << 25));
    g1[1] = (int)((width & 0xFFFFu) << 16);               // tensor_dim0 lo
    g1[2] = (int)((width >> 16) | ((rows & 0xFFFFu) << 16)); // td0 hi | td1 lo
    g1[3] = (int)((rows >> 16) | ((width & 0xFFFFu) << 16)); // td1 hi | tile_dim0
    g1[4] = (int)(rows & 0xFFFFu);                // tile_dim1, tile_dim2=0
    g1[5] = (int)stride;                          // tensor_dim0_stride lo
    g1[6] = 0;
    g1[7] = 0;
    i32x4 gz4 = {0, 0, 0, 0};
    i32x8 gz8 = {0, 0, 0, 0, 0, 0, 0, 0};
    __builtin_amdgcn_tensor_load_to_lds(g0, g1, gz4, gz4, gz8, 0);
}
#endif

// ---- kernel 1: x [B,C,H,W] f32 -> xr [B,S,C] bf16 (LDS tile transpose) ------
// grid = (S/32, C/32, B); block = 256. Coalesced on both sides.
__global__ __launch_bounds__(256) void k_prep_x(const float* __restrict__ x,
                                                __bf16* __restrict__ xr) {
    __shared__ float t[32][33];
    const int b  = blockIdx.z;
    const int s0 = blockIdx.x * 32, c0 = blockIdx.y * 32;
    const int tx = threadIdx.x & 31;       // fast index
    const int ty = threadIdx.x >> 5;       // 8 slow rows
#pragma unroll
    for (int i = 0; i < 4; ++i) {          // read coalesced in s
        int c = c0 + ty + i * 8;
        t[ty + i * 8][tx] = x[((size_t)b * C_ + c) * S_ + s0 + tx];
    }
    __syncthreads();
#pragma unroll
    for (int i = 0; i < 4; ++i) {          // write coalesced in c
        int s = s0 + ty + i * 8;
        xr[((size_t)b * S_ + s) * C_ + c0 + tx] = (__bf16)t[tx][ty + i * 8];
    }
}

// ---- kernel 2: w [C,d] f32 -> wT [d,C] bf16 (x4, LDS tile transpose) --------
// grid = (C/32, C/32, 4); block = 256.
__global__ __launch_bounds__(256) void k_prep_w(const float* __restrict__ wq,
                                                const float* __restrict__ wk,
                                                const float* __restrict__ wv,
                                                const float* __restrict__ wo,
                                                __bf16* wqT, __bf16* wkT,
                                                __bf16* wvT, __bf16* woT) {
    __shared__ float t[32][33];
    const float* src; __bf16* dst;
    switch (blockIdx.z) {
        case 0:  src = wq; dst = wqT; break;
        case 1:  src = wk; dst = wkT; break;
        case 2:  src = wv; dst = wvT; break;
        default: src = wo; dst = woT; break;
    }
    const int k0 = blockIdx.x * 32, n0 = blockIdx.y * 32;
    const int tx = threadIdx.x & 31;
    const int ty = threadIdx.x >> 5;
#pragma unroll
    for (int i = 0; i < 4; ++i) {          // read w[k][n] coalesced in n
        int kk = k0 + ty + i * 8;
        t[ty + i * 8][tx] = src[(size_t)kk * C_ + n0 + tx];
    }
    __syncthreads();
#pragma unroll
    for (int i = 0; i < 4; ++i) {          // write wT[n][k] coalesced in k
        int n = n0 + ty + i * 8;
        dst[(size_t)n * C_ + k0 + tx] = (__bf16)t[tx][ty + i * 8];
    }
}

// ---- kernel 3: QKV projection GEMM ------------------------------------------
// grid = (8 col groups of 64, 64 row groups of 128, 3 matrices); block = 256.
// Each wave: 16x64 output (4 acc tiles, A reused 4x). Register double-buffered:
// iteration kt+1's fragments are loaded before kt's WMMAs issue.
__global__ __launch_bounds__(256) void k_qkv(const __bf16* __restrict__ xr,
                                             const __bf16* __restrict__ wqT,
                                             const __bf16* __restrict__ wkT,
                                             const __bf16* __restrict__ wvT,
                                             const float* __restrict__ bq,
                                             const float* __restrict__ bk,
                                             const float* __restrict__ bv,
                                             __bf16* __restrict__ q,
                                             __bf16* __restrict__ k,
                                             __bf16* __restrict__ vt) {
    const int mat = blockIdx.z;
    const __bf16* wT  = (mat == 0) ? wqT : (mat == 1) ? wkT : wvT;
    const float* bias = (mat == 0) ? bq  : (mat == 1) ? bk  : bv;

    const int wave = threadIdx.x >> 5, lane = threadIdx.x & 31;
    const int lr = lane & 15, kh = lane >> 4;
    const int n0   = blockIdx.x * 64;
    const int row0 = blockIdx.y * 128 + wave * 16;

    v8f acc0 = {}, acc1 = {}, acc2 = {}, acc3 = {};
    const __bf16* ap = xr + (size_t)(row0 + lr) * C_ + kh * 8;
    const __bf16* bp = wT + (size_t)(n0 + lr) * C_ + kh * 8;

    v16bf a  = load_bf16_frag(ap);
    v16bf b0 = load_bf16_frag(bp);
    v16bf b1 = load_bf16_frag(bp + 16 * C_);
    v16bf b2 = load_bf16_frag(bp + 32 * C_);
    v16bf b3 = load_bf16_frag(bp + 48 * C_);
    for (int kt = 0; kt < 15; ++kt) {
        const __bf16* ap2 = ap + (kt + 1) * 32;
        const __bf16* bp2 = bp + (kt + 1) * 32;
        v16bf na  = load_bf16_frag(ap2);
        v16bf nb0 = load_bf16_frag(bp2);
        v16bf nb1 = load_bf16_frag(bp2 + 16 * C_);
        v16bf nb2 = load_bf16_frag(bp2 + 32 * C_);
        v16bf nb3 = load_bf16_frag(bp2 + 48 * C_);
        acc0 = wmma_bf16(a, b0, acc0);
        acc1 = wmma_bf16(a, b1, acc1);
        acc2 = wmma_bf16(a, b2, acc2);
        acc3 = wmma_bf16(a, b3, acc3);
        a = na; b0 = nb0; b1 = nb1; b2 = nb2; b3 = nb3;
    }
    acc0 = wmma_bf16(a, b0, acc0);
    acc1 = wmma_bf16(a, b1, acc1);
    acc2 = wmma_bf16(a, b2, acc2);
    acc3 = wmma_bf16(a, b3, acc3);

#pragma unroll
    for (int t = 0; t < 4; ++t) {
        v8f acc = (t == 0) ? acc0 : (t == 1) ? acc1 : (t == 2) ? acc2 : acc3;
        const int col = n0 + t * 16 + lr;
        const float bval = bias[col];
        if (mat < 2) {
            __bf16* out = (mat == 0) ? q : k;
#pragma unroll
            for (int r = 0; r < 8; ++r) {
                int m = r + 8 * kh;
                out[(size_t)(row0 + m) * C_ + col] = (__bf16)(acc[r] + bval);
            }
        } else {                // store V transposed: vt[b][h][d][s]
            const int h = col >> 6, d = col & 63;
#pragma unroll
            for (int r = 0; r < 8; ++r) {
                int grow = row0 + r + 8 * kh;
                int b = grow >> 10, s = grow & (S_ - 1);
                vt[(((size_t)b * NH + h) * HD + d) * S_ + s] = (__bf16)(acc[r] + bval);
            }
        }
    }
}

// ---- kernel 4: fused flash attention ----------------------------------------
// grid = B*NH*8 = 512 blocks; block = 256 (8 waves); wave = 16 query rows.
// Double-buffered K/V tiles staged by the Tensor Data Mover: while all 8 waves
// compute on buffer `cur`, wave 0 has the TDM filling buffer `cur^1` for the
// next 32-key block (s_wait_tensorcnt(2) waits only for the current pair).
#define KPAD 72   // 64 + 8 bf16 pad  (row = 32 dwords, pad 4 dwords)
#define VPAD 40   // 32 + 8 bf16 pad  (row = 16 dwords, pad 4 dwords)
#define PPAD 40
__global__ __launch_bounds__(256) void k_attn(const __bf16* __restrict__ q,
                                              const __bf16* __restrict__ kmat,
                                              const __bf16* __restrict__ vt,
                                              __bf16* __restrict__ attn_out) {
    __shared__ __bf16 ldsK[2][32 * KPAD];    // [key][d]   2 x 4.5 KB
    __shared__ __bf16 ldsV[2][64 * VPAD];    // [d][key]   2 x 5   KB
    __shared__ __bf16 ldsP[8][16 * PPAD];    // per-wave P 10  KB

    const int tid  = threadIdx.x;
    const int wave = tid >> 5, lane = tid & 31;
    const int lr = lane & 15, kh = lane >> 4;
    const int bh = blockIdx.x >> 3;
    const int qc = blockIdx.x & 7;
    const int b = bh >> 3, h = bh & 7;
    const int q0 = qc * 128 + wave * 16;
    const float scale = 0.04419417382415922f;   // 1/sqrt(512)

    const __bf16* qp = q + (size_t)(b * S_ + q0 + lr) * C_ + h * HD + kh * 8;
    const v16bf qa0 = load_bf16_frag(qp);
    const v16bf qa1 = load_bf16_frag(qp + 32);

    float m[8], l[8];
    v8f o0 = {}, o1 = {}, o2 = {}, o3 = {};
#pragma unroll
    for (int r = 0; r < 8; ++r) { m[r] = -1e30f; l[r] = 0.0f; }

    const __bf16* kbase = kmat + (size_t)(b * S_) * C_ + h * HD;      // + key*512
    const __bf16* vbase = vt + ((size_t)b * NH + h) * HD * S_;        // + d*1024

#ifdef USE_TDM
    if (wave == 0) {
        tdm_load_2d(lds_off(&ldsK[0][0]), kbase, 64u, 32u, (unsigned)C_, 4u, 3u);
        tdm_load_2d(lds_off(&ldsV[0][0]), vbase, 32u, 64u, (unsigned)S_, 3u, 3u);
    }
#endif

    for (int kb = 0; kb < 32; ++kb) {
        const int cur = kb & 1;
        __syncthreads();      // readers of the buffer being refilled are done
#ifdef USE_TDM
        if (wave == 0) {
            if (kb + 1 < 32) {
                const int n1 = (kb + 1) * 32;
                tdm_load_2d(lds_off(&ldsK[cur ^ 1][0]), kbase + (size_t)n1 * C_,
                            64u, 32u, (unsigned)C_, 4u, 3u);
                tdm_load_2d(lds_off(&ldsV[cur ^ 1][0]), vbase + n1,
                            32u, 64u, (unsigned)S_, 3u, 3u);
                __builtin_amdgcn_s_wait_tensorcnt(2);   // current pair done
            } else {
                __builtin_amdgcn_s_wait_tensorcnt(0);
            }
        }
#else
        {   // cooperative fallback: 256 x 16B each for K and V
            const int n0 = kb * 32;
            int key = tid >> 3, ck = tid & 7;
            *(v8bf*)&ldsK[cur][key * KPAD + ck * 8] =
                *(const v8bf*)(kbase + (size_t)(n0 + key) * C_ + ck * 8);
            int d = tid >> 2, cv = tid & 3;
            *(v8bf*)&ldsV[cur][d * VPAD + cv * 8] =
                *(const v8bf*)(vbase + (size_t)d * S_ + n0 + cv * 8);
        }
#endif
        __syncthreads();                       // staged tiles visible

        // energy: Q(16x64) x K^T -> two 16x16 tiles
        const __bf16* kf0 = &ldsK[cur][lr * KPAD + kh * 8];
        const __bf16* kf1 = &ldsK[cur][(16 + lr) * KPAD + kh * 8];
        v8f s0 = {}, s1 = {};
        s0 = wmma_bf16(qa0, load_bf16_frag(kf0),      s0);
        s0 = wmma_bf16(qa1, load_bf16_frag(kf0 + 32), s0);
        s1 = wmma_bf16(qa0, load_bf16_frag(kf1),      s1);
        s1 = wmma_bf16(qa1, load_bf16_frag(kf1 + 32), s1);

        // online softmax per row (row lives across 16 lanes)
#pragma unroll
        for (int r = 0; r < 8; ++r) {
            float e0 = s0[r] * scale, e1 = s1[r] * scale;
            float mx = fmaxf(e0, e1);
#pragma unroll
            for (int off = 1; off < 16; off <<= 1)
                mx = fmaxf(mx, __shfl_xor(mx, off, 32));
            float mnew = fmaxf(m[r], mx);
            float p0 = __expf(e0 - mnew);
            float p1 = __expf(e1 - mnew);
            float rs = p0 + p1;
#pragma unroll
            for (int off = 1; off < 16; off <<= 1)
                rs += __shfl_xor(rs, off, 32);
            float f = __expf(m[r] - mnew);
            l[r] = l[r] * f + rs;
            m[r] = mnew;
            o0[r] *= f; o1[r] *= f; o2[r] *= f; o3[r] *= f;
            int rowm = r + 8 * kh;
            ldsP[wave][rowm * PPAD + lr]      = (__bf16)p0;
            ldsP[wave][rowm * PPAD + 16 + lr] = (__bf16)p1;
        }
        // reload P as A-fragment (same wave; hw keeps DS ops in order)
        v16bf pa = load_bf16_frag(&ldsP[wave][lr * PPAD + kh * 8]);

        // P(16x32) x V(32x64): 4 N-tiles from ldsV
        o0 = wmma_bf16(pa, load_bf16_frag(&ldsV[cur][lr * VPAD + kh * 8]),        o0);
        o1 = wmma_bf16(pa, load_bf16_frag(&ldsV[cur][(16 + lr) * VPAD + kh * 8]), o1);
        o2 = wmma_bf16(pa, load_bf16_frag(&ldsV[cur][(32 + lr) * VPAD + kh * 8]), o2);
        o3 = wmma_bf16(pa, load_bf16_frag(&ldsV[cur][(48 + lr) * VPAD + kh * 8]), o3);
    }

    // epilogue: divide by row sums, store bf16 [B,S,512]
#pragma unroll
    for (int r = 0; r < 8; ++r) {
        float inv = 1.0f / l[r];
        int sidx = q0 + r + 8 * kh;
        __bf16* op = attn_out + (size_t)(b * S_ + sidx) * C_ + h * HD + lr;
        op[0]  = (__bf16)(o0[r] * inv);
        op[16] = (__bf16)(o1[r] * inv);
        op[32] = (__bf16)(o2[r] * inv);
        op[48] = (__bf16)(o3[r] * inv);
    }
}

// ---- kernel 5: output projection + bias + residual + NCHW store -------------
__global__ __launch_bounds__(256) void k_out(const __bf16* __restrict__ attn_out,
                                             const __bf16* __restrict__ woT,
                                             const float* __restrict__ bo,
                                             const float* __restrict__ x,
                                             float* __restrict__ y) {
    const int wave = threadIdx.x >> 5, lane = threadIdx.x & 31;
    const int lr = lane & 15, kh = lane >> 4;
    const int n0   = blockIdx.x * 64;
    const int row0 = blockIdx.y * 128 + wave * 16;

    v8f acc0 = {}, acc1 = {}, acc2 = {}, acc3 = {};
    const __bf16* ap = attn_out + (size_t)(row0 + lr) * C_ + kh * 8;
    const __bf16* bp = woT      + (size_t)(n0 + lr) * C_ + kh * 8;

    v16bf a  = load_bf16_frag(ap);
    v16bf b0 = load_bf16_frag(bp);
    v16bf b1 = load_bf16_frag(bp + 16 * C_);
    v16bf b2 = load_bf16_frag(bp + 32 * C_);
    v16bf b3 = load_bf16_frag(bp + 48 * C_);
    for (int kt = 0; kt < 15; ++kt) {
        const __bf16* ap2 = ap + (kt + 1) * 32;
        const __bf16* bp2 = bp + (kt + 1) * 32;
        v16bf na  = load_bf16_frag(ap2);
        v16bf nb0 = load_bf16_frag(bp2);
        v16bf nb1 = load_bf16_frag(bp2 + 16 * C_);
        v16bf nb2 = load_bf16_frag(bp2 + 32 * C_);
        v16bf nb3 = load_bf16_frag(bp2 + 48 * C_);
        acc0 = wmma_bf16(a, b0, acc0);
        acc1 = wmma_bf16(a, b1, acc1);
        acc2 = wmma_bf16(a, b2, acc2);
        acc3 = wmma_bf16(a, b3, acc3);
        a = na; b0 = nb0; b1 = nb1; b2 = nb2; b3 = nb3;
    }
    acc0 = wmma_bf16(a, b0, acc0);
    acc1 = wmma_bf16(a, b1, acc1);
    acc2 = wmma_bf16(a, b2, acc2);
    acc3 = wmma_bf16(a, b3, acc3);

#pragma unroll
    for (int t = 0; t < 4; ++t) {
        v8f acc = (t == 0) ? acc0 : (t == 1) ? acc1 : (t == 2) ? acc2 : acc3;
        const int c = n0 + t * 16 + lr;
        const float bval = bo[c];
#pragma unroll
        for (int r = 0; r < 8; ++r) {
            int grow = row0 + r + 8 * kh;
            int b = grow >> 10, s = grow & (S_ - 1);
            size_t oi = ((size_t)b * C_ + c) * S_ + s;
            y[oi] = acc[r] + bval + x[oi];
        }
    }
}

// ---- host launcher ----------------------------------------------------------
extern "C" void kernel_launch(void* const* d_in, const int* in_sizes, int n_in,
                              void* d_out, int out_size, void* d_ws, size_t ws_size,
                              hipStream_t stream) {
    const float* x  = (const float*)d_in[0];
    const float* wq = (const float*)d_in[1];
    const float* bq = (const float*)d_in[2];
    const float* wk = (const float*)d_in[3];
    const float* bk = (const float*)d_in[4];
    const float* wv = (const float*)d_in[5];
    const float* bv = (const float*)d_in[6];
    const float* wo = (const float*)d_in[7];
    const float* bo = (const float*)d_in[8];
    float* y = (float*)d_out;

    char* ws = (char*)d_ws;
    const size_t SZ_ACT = (size_t)B_ * S_ * C_ * sizeof(__bf16);  // 8 MB
    const size_t SZ_W   = (size_t)C_ * C_ * sizeof(__bf16);       // 0.5 MB
    __bf16* xr   = (__bf16*)(ws);
    __bf16* wqT  = (__bf16*)(ws + SZ_ACT);
    __bf16* wkT  = (__bf16*)(ws + SZ_ACT + 1 * SZ_W);
    __bf16* wvT  = (__bf16*)(ws + SZ_ACT + 2 * SZ_W);
    __bf16* woT  = (__bf16*)(ws + SZ_ACT + 3 * SZ_W);
    __bf16* qb   = (__bf16*)(ws + SZ_ACT + 4 * SZ_W);
    __bf16* kb   = (__bf16*)(ws + 2 * SZ_ACT + 4 * SZ_W);
    __bf16* vtb  = (__bf16*)(ws + 3 * SZ_ACT + 4 * SZ_W);
    __bf16* attn = (__bf16*)(ws + 4 * SZ_ACT + 4 * SZ_W);

    dim3 gx(S_ / 32, C_ / 32, B_);
    k_prep_x<<<gx, 256, 0, stream>>>(x, xr);
    dim3 gw(C_ / 32, C_ / 32, 4);
    k_prep_w<<<gw, 256, 0, stream>>>(wq, wk, wv, wo, wqT, wkT, wvT, woT);
    dim3 gq(C_ / 64, (B_ * S_) / 128, 3);
    k_qkv<<<gq, 256, 0, stream>>>(xr, wqT, wkT, wvT, bq, bk, bv, qb, kb, vtb);
    k_attn<<<B_ * NH * 8, 256, 0, stream>>>(qb, kb, vtb, attn);
    dim3 go(C_ / 64, (B_ * S_) / 128);
    k_out<<<go, 256, 0, stream>>>(attn, woT, bo, x, y);
}
